// MHA_18657337934739
// MI455X (gfx1250) — matrix operand order repaired and checked by reading, not compiled, not run on hardware
//
#include <hip/hip_runtime.h>
#include <hip/hip_bf16.h>

typedef __bf16 bf16;
typedef bf16  v16bf __attribute__((ext_vector_type(16)));
typedef bf16  v8bf  __attribute__((ext_vector_type(8)));
typedef float v8f   __attribute__((ext_vector_type(8)));

#define B_  4
#define T_  2048
#define C_  1024
#define H_  16
#define D_  64
#define M_  (B_ * T_)          // 8192 rows of x
#define N_QKV (3 * C_)         // 3072

__device__ __forceinline__ v8f vzero8() {
  v8f v;
#pragma unroll
  for (int i = 0; i < 8; ++i) v[i] = 0.0f;
  return v;
}

__device__ __forceinline__ v8f wmma_bf16(v16bf a, v16bf b, v8f c) {
  return __builtin_amdgcn_wmma_f32_16x16x32_bf16(false, a, false, b, (short)0, c,
                                                 false, false);
}

// Reduce across the 16 lanes of each half-wave (C/D rows live per (vgpr, half)).
__device__ __forceinline__ float half16_max(float v) {
#pragma unroll
  for (int m = 1; m < 16; m <<= 1) v = fmaxf(v, __shfl_xor(v, m, 32));
  return v;
}
__device__ __forceinline__ float half16_sum(float v) {
#pragma unroll
  for (int m = 1; m < 16; m <<= 1) v += __shfl_xor(v, m, 32);
  return v;
}

// -------- elementwise f32 -> bf16 --------
__global__ void cvt_bf16_kernel(const float* __restrict__ in, bf16* __restrict__ out, int n) {
  int i = blockIdx.x * blockDim.x + threadIdx.x;
  if (i < n) out[i] = (bf16)in[i];
}

// -------- f32 [R][N] -> bf16 [N][R] (transpose-convert for WMMA B operand) --------
__global__ void cvt_transpose_kernel(const float* __restrict__ in, bf16* __restrict__ out,
                                     int R, int N) {
  int i = blockIdx.x * blockDim.x + threadIdx.x;
  if (i < R * N) {
    int r = i / N, c = i % N;
    out[(size_t)c * R + r] = (bf16)in[i];
  }
}

// Load one K=32 stage of A (16x32) and B (32x64 as 4 subtiles) operands.
__device__ __forceinline__ void load_ab(const bf16* __restrict__ arow,
                                        const bf16* __restrict__ bcol,
                                        int K, int k0, v16bf& a, v16bf b[4]) {
  // A k order used here (contiguous, as B): halves [0..7] = K (16h..16h+7), halves [8..15] = K (16h+8..16h+15)
  v8bf lo = *(const v8bf*)(arow + k0);
  v8bf hi = *(const v8bf*)(arow + k0 + 8);
#pragma unroll
  for (int i = 0; i < 8; ++i) { a[i] = lo[i]; a[i + 8] = hi[i]; }
#pragma unroll
  for (int ns = 0; ns < 4; ++ns)
    b[ns] = *(const v16bf*)(bcol + (size_t)ns * 16 * K + k0);
}

// -------- per-wave 16(M) x 64(N) GEMM tile, bf16 in, f32 accum ----------------
// Ping-pong register pipelining (no cross-stage copies): load stage s+1 while
// the stage-s WMMAs drain.  Requires K multiple of 64 and K >= 64.
// A: [M][K] row-major bf16.  Bt: [N][K] row-major bf16 (i.e. B transposed).
__device__ __forceinline__ void gemm_tile_16x64(const bf16* __restrict__ A,
                                                const bf16* __restrict__ Bt,
                                                int K, int m0, int n0, v8f acc[4]) {
  const int lane = threadIdx.x & 31;
  const int l15  = lane & 15;
  const int hh   = lane >> 4;
  const bf16* arow = A  + (size_t)(m0 + l15) * K + 16 * hh;
  const bf16* bcol = Bt + (size_t)(n0 + l15) * K + 16 * hh;

  v16bf aA, aB, bA[4], bB[4];
  load_ab(arow, bcol, K, 0, aA, bA);
  int k0 = 0;
  for (; k0 + 64 < K; k0 += 64) {
    load_ab(arow, bcol, K, k0 + 32, aB, bB);
#pragma unroll
    for (int ns = 0; ns < 4; ++ns) acc[ns] = wmma_bf16(aA, bA[ns], acc[ns]);
    load_ab(arow, bcol, K, k0 + 64, aA, bA);
#pragma unroll
    for (int ns = 0; ns < 4; ++ns) acc[ns] = wmma_bf16(aB, bB[ns], acc[ns]);
  }
  load_ab(arow, bcol, K, k0 + 32, aB, bB);
#pragma unroll
  for (int ns = 0; ns < 4; ++ns) acc[ns] = wmma_bf16(aA, bA[ns], acc[ns]);
#pragma unroll
  for (int ns = 0; ns < 4; ++ns) acc[ns] = wmma_bf16(aB, bB[ns], acc[ns]);
}

// -------- QKV GEMM + bias + RoPE + scatter to [B,H,T,D] (q,k) / [B,H,D,T] (v) --------
__global__ void __launch_bounds__(256)
qkv_rope_kernel(const bf16* __restrict__ xb, const bf16* __restrict__ wqkvT,
                const float* __restrict__ bias,
                bf16* __restrict__ qb, bf16* __restrict__ kbuf, bf16* __restrict__ vT) {
  const int wave = threadIdx.x >> 5;
  const int lane = threadIdx.x & 31;
  const int l15 = lane & 15, hh = lane >> 4;
  const int tile = blockIdx.x * 8 + wave;            // 24576 wave tiles
  const int tn = tile % (N_QKV / 64);                // 48 N-tiles
  const int tm = tile / (N_QKV / 64);
  const int m0 = tm * 16, n0 = tn * 64;

  v8f acc[4];
#pragma unroll
  for (int i = 0; i < 4; ++i) acc[i] = vzero8();
  gemm_tile_16x64(xb, wqkvT, C_, m0, n0, acc);

  const int reg  = n0 >> 10;                         // 0=q 1=k 2=v (tile never straddles)
  const int hcol = (n0 & 1023) >> 6;                 // head index (tile is one head wide)
  const float KL = 0.41524101186092029f;             // log2(10000)/32

#pragma unroll
  for (int ns = 0; ns < 4; ++ns) {
    const int d = ns * 16 + l15;
    const float bv = bias[n0 + d];
    const int p = d >> 1;
    const float freq = exp2f(-(float)p * KL);
#pragma unroll
    for (int r = 0; r < 8; ++r) {
      const int row = m0 + r + 8 * hh;               // global M index = b*T + t
      const int bidx = row >> 11;
      const int t = row & (T_ - 1);
      float val = acc[ns][r] + bv;
      if (reg == 2) {
        vT[((size_t)(bidx * H_ + hcol) * D_ + d) * T_ + t] = (bf16)val;
      } else {
        // interleaved-pair RoPE: partner element is the adjacent lane (same row)
        float partner = __shfl_xor(val, 1, 32);
        float th = (float)t * freq;
        float sn = __sinf(th);
        float cs = __cosf(th);
        float rot = val * cs + partner * sn * ((d & 1) ? 1.0f : -1.0f);
        size_t idx = ((size_t)(bidx * H_ + hcol) * T_ + t) * D_ + d;
        if (reg == 0) qb[idx] = (bf16)(rot * 0.125f); // fold 1/sqrt(D) into q
        else          kbuf[idx] = (bf16)rot;
      }
    }
  }
}

// -------- flash attention: one wave per 16-query tile, 32-key blocks ----------
// All 8 waves of a block share (b,h): the 32-key x 64-d K-tile (4KB, contiguous)
// is staged into a double-buffered LDS buffer by wave 0 with the gfx1250 async
// DMA path (global_load_async_to_lds_b128 + s_wait_asynccnt), published with a
// block barrier, and consumed by every wave via DS reads -> 8x less L2 traffic
// and the copy overlaps a full iteration of WMMA + softmax work.
__global__ void __launch_bounds__(256)
attn_kernel(const bf16* __restrict__ qb, const bf16* __restrict__ kb,
            const bf16* __restrict__ vT, bf16* __restrict__ attn_out) {
  __shared__ bf16 Kst[2][32 * 64];                    // 2 x 4KB K-tile double buffer
  __shared__ bf16 Pbuf[8 * 512];                      // 1KB per wave (16x32 bf16)
  const int wave = threadIdx.x >> 5;
  const int lane = threadIdx.x & 31;
  const int l15 = lane & 15, hh = lane >> 4;
  const int bh = blockIdx.x >> 4;                     // T/128 = 16 q-chunks per (b,h)
  const int qblk = blockIdx.x & 15;
  const int q0 = qblk * 128 + wave * 16;
  const int b = bh >> 4, head = bh & 15;

  const bf16* q  = qb + (size_t)bh * T_ * D_;
  const bf16* kk = kb + (size_t)bh * T_ * D_;
  const bf16* vt = vT + (size_t)bh * D_ * T_;

  // q A-operands (resident across the whole key loop); K-dim = D = 64 -> 2 chunks
  const bf16* qrow = q + (size_t)(q0 + l15) * D_;
  v16bf qa[2];
#pragma unroll
  for (int c = 0; c < 2; ++c) {
    v8bf lo = *(const v8bf*)(qrow + 32 * c + 16 * hh);
    v8bf hi = *(const v8bf*)(qrow + 32 * c + 16 * hh + 8);
#pragma unroll
    for (int i = 0; i < 8; ++i) { qa[c][i] = lo[i]; qa[c][i + 8] = hi[i]; }
  }

  v8f o[4];
#pragma unroll
  for (int i = 0; i < 4; ++i) o[i] = vzero8();
  float m[8], l[8];
#pragma unroll
  for (int r = 0; r < 8; ++r) { m[r] = -1e30f; l[r] = 0.0f; }

  bf16* pb = Pbuf + wave * 512;

  // async-stage key block 0 into buffer 0
  if (wave == 0) {
    const bf16* gsrc = kk;                                     // 4KB contiguous
    unsigned ldsbase = (unsigned)(size_t)(&Kst[0][0]);
#pragma unroll
    for (int i = 0; i < 8; ++i) {
      unsigned off = (unsigned)((i * 32 + lane) * 16);
      asm volatile("global_load_async_to_lds_b128 %0, %1, %2"
                   :
                   : "v"(ldsbase + off), "v"(off), "s"(gsrc)
                   : "memory");
    }
    asm volatile("s_wait_asynccnt 0x0" ::: "memory");
  }
  __syncthreads();

  int cur = 0;
  for (int kb0 = 0; kb0 < T_; kb0 += 32) {
    // kick off DMA of the next key block into the other buffer
    if (wave == 0) {
      int kn = (kb0 + 32 < T_) ? (kb0 + 32) : 0;
      const bf16* gsrc = kk + (size_t)kn * D_;
      unsigned ldsbase = (unsigned)(size_t)(&Kst[cur ^ 1][0]);
#pragma unroll
      for (int i = 0; i < 8; ++i) {
        unsigned off = (unsigned)((i * 32 + lane) * 16);
        asm volatile("global_load_async_to_lds_b128 %0, %1, %2"
                     :
                     : "v"(ldsbase + off), "v"(off), "s"(gsrc)
                     : "memory");
      }
    }

    const bf16* kst = &Kst[cur][0];                  // [32 keys][64 d]
    v8f s0 = vzero8(), s1 = vzero8();
#pragma unroll
    for (int c = 0; c < 2; ++c) {
      v16bf bk0 = *(const v16bf*)(kst + (size_t)l15 * D_ + 32 * c + 16 * hh);
      v16bf bk1 = *(const v16bf*)(kst + (size_t)(16 + l15) * D_ + 32 * c + 16 * hh);
      s0 = wmma_bf16(qa[c], bk0, s0);
      s1 = wmma_bf16(qa[c], bk1, s1);
    }

    // issue V loads early so they overlap the softmax VALU work below
    v16bf bv[4];
#pragma unroll
    for (int ds = 0; ds < 4; ++ds)
      bv[ds] = *(const v16bf*)(vt + (size_t)(ds * 16 + l15) * T_ + kb0 + 16 * hh);

    // online softmax, per C/D row r (row index = r + 8*half)
#pragma unroll
    for (int r = 0; r < 8; ++r) {
      float rmax = half16_max(fmaxf(s0[r], s1[r]));
      float mn = fmaxf(m[r], rmax);
      float alpha = __expf(m[r] - mn);
      m[r] = mn;
      float p0 = __expf(s0[r] - mn);
      float p1 = __expf(s1[r] - mn);
      l[r] = l[r] * alpha + half16_sum(p0 + p1);
#pragma unroll
      for (int ds = 0; ds < 4; ++ds) o[ds][r] *= alpha;
      pb[(r + 8 * hh) * 32 + l15]      = (bf16)p0;   // LDS transpose C/D -> A layout
      pb[(r + 8 * hh) * 32 + 16 + l15] = (bf16)p1;
    }
    asm volatile("s_wait_dscnt 0x0" ::: "memory");

    v8bf plo = *(const v8bf*)(pb + l15 * 32 + 16 * hh);
    v8bf phi = *(const v8bf*)(pb + l15 * 32 + 16 * hh + 8);
    v16bf pa;
#pragma unroll
    for (int i = 0; i < 8; ++i) { pa[i] = plo[i]; pa[i + 8] = phi[i]; }

#pragma unroll
    for (int ds = 0; ds < 4; ++ds)
      o[ds] = wmma_bf16(pa, bv[ds], o[ds]);          // V^T [D][T]: contiguous keys

    // make the freshly DMA'd buffer visible to all waves
    if (wave == 0) asm volatile("s_wait_asynccnt 0x0" ::: "memory");
    __syncthreads();
    cur ^= 1;
  }

  // normalize and scatter to [B*T, C] bf16 for the proj GEMM
#pragma unroll
  for (int r = 0; r < 8; ++r) {
    float inv = 1.0f / l[r];
    int t = q0 + r + 8 * hh;
#pragma unroll
    for (int ds = 0; ds < 4; ++ds) {
      attn_out[(size_t)(b * T_ + t) * C_ + head * D_ + ds * 16 + l15] =
          (bf16)(o[ds][r] * inv);
    }
  }
}

// -------- output projection + bias (f32 out) --------
__global__ void __launch_bounds__(256)
proj_kernel(const bf16* __restrict__ attn, const bf16* __restrict__ wprojT,
            const float* __restrict__ bias, float* __restrict__ out) {
  const int wave = threadIdx.x >> 5;
  const int lane = threadIdx.x & 31;
  const int l15 = lane & 15, hh = lane >> 4;
  const int tile = blockIdx.x * 8 + wave;            // 8192 wave tiles
  const int tn = tile & 15;                          // C/64 = 16
  const int tm = tile >> 4;
  const int m0 = tm * 16, n0 = tn * 64;

  v8f acc[4];
#pragma unroll
  for (int i = 0; i < 4; ++i) acc[i] = vzero8();
  gemm_tile_16x64(attn, wprojT, C_, m0, n0, acc);

#pragma unroll
  for (int ns = 0; ns < 4; ++ns) {
    const int n = n0 + ns * 16 + l15;
    const float bv = bias[n];
#pragma unroll
    for (int r = 0; r < 8; ++r) {
      out[(size_t)(m0 + r + 8 * hh) * C_ + n] = acc[ns][r] + bv;
    }
  }
}

extern "C" void kernel_launch(void* const* d_in, const int* in_sizes, int n_in,
                              void* d_out, int out_size, void* d_ws, size_t ws_size,
                              hipStream_t stream) {
  const float* x      = (const float*)d_in[0];
  const float* w_qkv  = (const float*)d_in[1];
  const float* b_qkv  = (const float*)d_in[2];
  const float* w_proj = (const float*)d_in[3];
  const float* b_proj = (const float*)d_in[4];
  float* out = (float*)d_out;

  // workspace layout (bytes)
  char* ws = (char*)d_ws;
  bf16* xb      = (bf16*)(ws);                                  // 8192*1024      -> 16 MB
  bf16* wqkvT   = (bf16*)(ws + (size_t)16 * 1024 * 1024);       // 3072*1024      ->  6 MB
  bf16* wprojT  = (bf16*)(ws + (size_t)22 * 1024 * 1024);       // 1024*1024      ->  2 MB
  bf16* qb      = (bf16*)(ws + (size_t)24 * 1024 * 1024);       // [B,H,T,D]      -> 16 MB
  bf16* kbuf    = (bf16*)(ws + (size_t)40 * 1024 * 1024);       // [B,H,T,D]      -> 16 MB
  bf16* vT      = (bf16*)(ws + (size_t)56 * 1024 * 1024);       // [B,H,D,T]      -> 16 MB
  bf16* attn_o  = (bf16*)(ws + (size_t)72 * 1024 * 1024);       // [B*T, C]       -> 16 MB

  // 1) precision conversion / weight transposition
  {
    int n = M_ * C_;
    cvt_bf16_kernel<<<(n + 255) / 256, 256, 0, stream>>>(x, xb, n);
  }
  {
    int n = C_ * N_QKV;
    cvt_transpose_kernel<<<(n + 255) / 256, 256, 0, stream>>>(w_qkv, wqkvT, C_, N_QKV);
  }
  {
    int n = C_ * C_;
    cvt_transpose_kernel<<<(n + 255) / 256, 256, 0, stream>>>(w_proj, wprojT, C_, C_);
  }

  // 2) QKV GEMM + bias + RoPE + layout scatter
  {
    int tiles = (M_ / 16) * (N_QKV / 64);       // 24576
    qkv_rope_kernel<<<tiles / 8, 256, 0, stream>>>(xb, wqkvT, b_qkv, qb, kbuf, vT);
  }

  // 3) attention (flash-style, one wave per 16 queries, async K staging)
  {
    int blocks = B_ * H_ * (T_ / 128);          // 1024
    attn_kernel<<<blocks, 256, 0, stream>>>(qb, kbuf, vT, attn_o);
  }

  // 4) output projection
  {
    int tiles = (M_ / 16) * (C_ / 64);          // 8192
    proj_kernel<<<tiles / 8, 256, 0, stream>>>(attn_o, wprojT, b_proj, out);
  }
}
